// TTTMLPLayer_58480274702562
// MI455X (gfx1250) — matrix operand order repaired and checked by reading, not compiled, not run on hardware
//
#include <hip/hip_runtime.h>
#include <hip/hip_bf16.h>

// ---------------------------------------------------------------------------
// TTT-MLP fused forward for MI455X (gfx1250, wave32, WMMA bf16 16x16x32).
//
// Roofline: 90 GFLOP, 200MB mandatory HBM traffic (8.6us @ 23.3TB/s) ->
// compute-bound at bf16 WMMA rates. Weights (2.75MB bf16) are L2-resident;
// 64-token tiles per workgroup bound L2 weight streaming to ~1.4GB total.
// ---------------------------------------------------------------------------

#define HID 768
#define INN 512
#define MT  64          // tokens per workgroup
#define EPSF 1e-5f

typedef unsigned short u16;
typedef u16   v8u  __attribute__((ext_vector_type(8)));
typedef u16   v16u __attribute__((ext_vector_type(16)));
typedef __bf16 v16bf __attribute__((ext_vector_type(16)));
typedef float  v8f  __attribute__((ext_vector_type(8)));

__device__ __forceinline__ u16 f2bf(float f) {
    unsigned int u = __builtin_bit_cast(unsigned int, f);
    u += 0x7fffu + ((u >> 16) & 1u);            // round-to-nearest-even
    return (u16)(u >> 16);
}

__device__ __forceinline__ v8f wmma_bf16(v16bf a, v16bf b, v8f c) {
    // D = A(16x32 bf16) x B(32x16 bf16) + C(16x16 f32)
    return __builtin_amdgcn_wmma_f32_16x16x32_bf16(
        /*neg_a=*/false, a, /*neg_b=*/false, b,
        /*c_mod=*/(short)0, c, /*reuse_a=*/false, /*reuse_b=*/false);
}

// B-fragment: column n of B == row n of weight [N,K]. The two 16-byte chunks
// at (kb + h*16) and (kb + h*16 + 8) give contiguous K = 16h + s, matching
// the A-fragment's contiguous-16 load.
__device__ __forceinline__ v16bf load_bfrag(const u16* wrow, int kb, int h) {
    const u16* p = wrow + kb + h * 16;
    v8u lo = *(const v8u*)(p);
    v8u hi = *(const v8u*)(p + 8);
    v16u bb;
#pragma unroll
    for (int j = 0; j < 8; ++j) { bb[j] = lo[j]; bb[j + 8] = hi[j]; }
    return __builtin_bit_cast(v16bf, bb);
}

// K-loop over one GEMM stage. A in LDS [rows][K] bf16 (row stride == K),
// W global bf16 [N][K]. acc[MTI][NTI] are 16x16 f32 tiles.
template <int K, int NTI, int MTI>
__device__ __forceinline__ void gemm_kloop(const u16* __restrict__ albase,
                                           const u16* __restrict__ wg,
                                           int nbase, int c, int h,
                                           v8f acc[MTI][NTI]) {
    for (int kb = 0; kb < K; kb += 32) {
        v16bf a[MTI];
#pragma unroll
        for (int mt = 0; mt < MTI; ++mt) {
            const u16* ap = albase + (mt * 16 + c) * K + kb + h * 16;
            a[mt] = __builtin_bit_cast(v16bf, *(const v16u*)ap);
        }
#pragma unroll
        for (int nt = 0; nt < NTI; ++nt) {
            const u16* wrow = wg + (size_t)(nbase + nt * 16 + c) * K;
            v16bf bf = load_bfrag(wrow, kb, h);
#pragma unroll
            for (int mt = 0; mt < MTI; ++mt)
                acc[mt][nt] = wmma_bf16(a[mt], bf, acc[mt][nt]);
        }
    }
}

// bias (+optional ReLU) epilogue, store bf16 tile into LDS [rows][dstride]
template <bool RELU, int NTI, int MTI>
__device__ __forceinline__ void epilogue_lds(v8f acc[MTI][NTI],
                                             const float* __restrict__ bias,
                                             u16* __restrict__ dst, int dstride,
                                             int nbase, int c, int h) {
#pragma unroll
    for (int nt = 0; nt < NTI; ++nt) {
        int n = nbase + nt * 16 + c;
        float bv = bias[n];
#pragma unroll
        for (int mt = 0; mt < MTI; ++mt)
#pragma unroll
            for (int v = 0; v < 8; ++v) {
                int r = mt * 16 + v + 8 * h;
                float t = acc[mt][nt][v] + bv;
                if (RELU) t = fmaxf(t, 0.f);
                dst[r * dstride + n] = f2bf(t);
            }
    }
}

__global__ void __launch_bounds__(256)
convert_weights_bf16(const float* __restrict__ W1, const float* __restrict__ W2,
                     const float* __restrict__ Wo, u16* __restrict__ ws) {
    const int i = blockIdx.x * 256 + threadIdx.x;
    const int N1 = INN * HID;            // 393216
    const int N2 = N1 + HID * INN;       // 786432
    const int N3 = N2 + HID * HID;       // 1376256
    if (i >= N3) return;
    float v;
    if (i < N1)      v = W1[i];
    else if (i < N2) v = W2[i - N1];
    else             v = Wo[i - N2];
    ws[i] = f2bf(v);
}

__global__ void __launch_bounds__(256)
ttt_mlp_fused(const float* __restrict__ x,
              const float* __restrict__ b1, const float* __restrict__ b2,
              const float* __restrict__ bo,
              const float* __restrict__ gamma, const float* __restrict__ beta,
              const u16* __restrict__ W1b, const u16* __restrict__ W2b,
              const u16* __restrict__ Wob, float* __restrict__ out) {
    // LDS: xa[64][768] bf16 (96KB) | hb[64][512] bf16 (64KB).
    // pred tile aliases xa; LN stats alias hb (only live after hb is dead).
    __shared__ __align__(32) unsigned char smem[98304 + 65536];
    u16*   xa    = (u16*)smem;                   // A for GEMM1 / pred for GEMM3
    u16*   hb    = (u16*)(smem + 98304);         // A for GEMM2
    float* rsum  = (float*)(smem + 98304);       // [8][64] per-wave row sums
    float* rsum2 = rsum + 512;                   // [8][64]
    float* smu   = rsum2 + 512;                  // [64]
    float* srs   = smu + 64;                     // [64]

    const int tid  = threadIdx.x;
    const int wave = tid >> 5;
    const int lane = tid & 31;
    const int h    = lane >> 4;                  // K-half selector
    const int c    = lane & 15;                  // row (A) / column (B,C)
    const int row0 = blockIdx.x * MT;

    // ---- Stage 0: x tile -> bf16 in LDS ------------------------------------
    {
        const float* xt = x + (size_t)row0 * HID;   // tile rows are contiguous
        for (int i = tid * 4; i < MT * HID; i += 256 * 4) {
            float4 v = *(const float4*)(xt + i);
            xa[i + 0] = f2bf(v.x); xa[i + 1] = f2bf(v.y);
            xa[i + 2] = f2bf(v.z); xa[i + 3] = f2bf(v.w);
        }
    }
    __syncthreads();

    // ---- Stage 1: h = relu(x @ W1^T + b1)  [64 x 512] ----------------------
    {
        v8f acc[4][4] = {};
        const int nb = wave * 64;
        gemm_kloop<HID, 4, 4>(xa, W1b, nb, c, h, acc);
        epilogue_lds<true, 4, 4>(acc, b1, hb, INN, nb, c, h);
    }
    __syncthreads();

    // ---- Stage 2: pred = h @ W2^T + b2  [64 x 768] (into xa-alias) ---------
    const int nb2 = wave * 96;
    {
        v8f acc[4][6] = {};
        gemm_kloop<INN, 6, 4>(hb, W2b, nb2, c, h, acc);
        epilogue_lds<false, 6, 4>(acc, b2, xa, HID, nb2, c, h);
    }
    __syncthreads();

    // ---- Stage 3: out = pred @ Wo^T + bo, fused LayerNorm ------------------
    {
        v8f acc[4][6] = {};
        gemm_kloop<HID, 6, 4>(xa, Wob, nb2, c, h, acc);

        // add output bias into accumulators
#pragma unroll
        for (int nt = 0; nt < 6; ++nt) {
            float bv = bo[nb2 + nt * 16 + c];
#pragma unroll
            for (int mt = 0; mt < 4; ++mt)
#pragma unroll
                for (int v = 0; v < 8; ++v) acc[mt][nt][v] += bv;
        }

        // per-row partial sum / sumsq: reduce 6 nt locally, then butterfly
        // across the 16 column-lanes of each half-wave (wave32 shuffles).
#pragma unroll
        for (int mt = 0; mt < 4; ++mt)
#pragma unroll
            for (int v = 0; v < 8; ++v) {
                float s = 0.f, s2 = 0.f;
#pragma unroll
                for (int nt = 0; nt < 6; ++nt) {
                    float t = acc[mt][nt][v];
                    s += t; s2 += t * t;
                }
#pragma unroll
                for (int m = 1; m < 16; m <<= 1) {
                    s  += __shfl_xor(s,  m, 32);
                    s2 += __shfl_xor(s2, m, 32);
                }
                if (c == 0) {
                    int r = mt * 16 + v + 8 * h;
                    rsum[wave * 64 + r]  = s;   // hb region is dead by now
                    rsum2[wave * 64 + r] = s2;
                }
            }
        __syncthreads();

        // deterministic cross-wave reduction (fixed order, no atomics)
        if (tid < 64) {
            float s = 0.f, s2 = 0.f;
#pragma unroll
            for (int w = 0; w < 8; ++w) { s += rsum[w * 64 + tid]; s2 += rsum2[w * 64 + tid]; }
            float mu  = s * (1.f / HID);
            float var = s2 * (1.f / HID) - mu * mu;
            smu[tid] = mu;
            srs[tid] = rsqrtf(var + EPSF);
        }
        __syncthreads();

        // normalize + affine + fp32 store
        float g[6], bt[6];
#pragma unroll
        for (int nt = 0; nt < 6; ++nt) {
            int n = nb2 + nt * 16 + c;
            g[nt] = gamma[n]; bt[nt] = beta[n];
        }
#pragma unroll
        for (int mt = 0; mt < 4; ++mt)
#pragma unroll
            for (int v = 0; v < 8; ++v) {
                int r = mt * 16 + v + 8 * h;
                float mu = smu[r], rs = srs[r];
                float* orow = out + (size_t)(row0 + r) * HID;
#pragma unroll
                for (int nt = 0; nt < 6; ++nt) {
                    int n = nb2 + nt * 16 + c;
                    orow[n] = (acc[mt][nt][v] - mu) * rs * g[nt] + bt[nt];
                }
            }
    }
}

extern "C" void kernel_launch(void* const* d_in, const int* in_sizes, int n_in,
                              void* d_out, int out_size, void* d_ws, size_t ws_size,
                              hipStream_t stream) {
    (void)n_in; (void)out_size; (void)ws_size;
    const float* x     = (const float*)d_in[0];
    const float* W1    = (const float*)d_in[1];
    const float* b1    = (const float*)d_in[2];
    const float* W2    = (const float*)d_in[3];
    const float* b2    = (const float*)d_in[4];
    const float* Wo    = (const float*)d_in[5];
    const float* bo    = (const float*)d_in[6];
    const float* gamma = (const float*)d_in[7];
    const float* beta  = (const float*)d_in[8];

    u16* ws  = (u16*)d_ws;
    u16* W1b = ws;                       // [512][768] bf16
    u16* W2b = ws + INN * HID;           // [768][512] bf16
    u16* Wob = W2b + HID * INN;          // [768][768] bf16

    const int wtotal = INN * HID + HID * INN + HID * HID;   // 1376256
    convert_weights_bf16<<<(wtotal + 255) / 256, 256, 0, stream>>>(W1, W2, Wo, ws);

    const int rows  = in_sizes[0] / HID;     // 32768 tokens
    const int tiles = rows / MT;             // 512 workgroups
    ttt_mlp_fused<<<tiles, 256, 0, stream>>>(x, b1, b2, bo, gamma, beta,
                                             W1b, W2b, Wob, (float*)d_out);
}
